// DSAttention_20272245637722
// MI455X (gfx1250) — compile-verified, matrix operand order checked by reading, not run on hardware
//
#include <hip/hip_runtime.h>
#include <hip/hip_bf16.h>

typedef __attribute__((ext_vector_type(16))) _Float16 v16h;
typedef __attribute__((ext_vector_type(8)))  float    v8f;

#define HW_FULL  65536   // 256*256
#define HW_QTR   16384   // 128*128
#define XSTR     80      // padded LDS row stride (halves): 160B, 32B-aligned

// Load one 32px x 64ch f32 subtile as float4s (coalesced), phase-split.
template <int HW>
__device__ __forceinline__ void tile_load(const float* __restrict__ X,
                                          int ps, int t, float4& va, float4& vb)
{
    const int p4 = (t & 7) * 4;
    const int c0 = t >> 3;                       // 0..31
    va = *(const float4*)&X[(size_t)c0 * HW + ps + p4];
    vb = *(const float4*)&X[(size_t)(c0 + 32) * HW + ps + p4];
    __builtin_prefetch(&X[(size_t)c0 * HW + ps + 64 + p4], 0, 1);
}

// Convert + scatter into LDS [pixel][channel] layout.
__device__ __forceinline__ void tile_store(_Float16* __restrict__ dst, int t,
                                           const float4& va, const float4& vb)
{
    const int p4 = (t & 7) * 4;
    const int c0 = t >> 3;
    _Float16* d0 = dst + p4 * XSTR + c0;
    d0[0 * XSTR] = (_Float16)va.x;  d0[1 * XSTR] = (_Float16)va.y;
    d0[2 * XSTR] = (_Float16)va.z;  d0[3 * XSTR] = (_Float16)va.w;
    _Float16* d1 = d0 + 32;
    d1[0 * XSTR] = (_Float16)vb.x;  d1[1 * XSTR] = (_Float16)vb.y;
    d1[2 * XSTR] = (_Float16)vb.z;  d1[3 * XSTR] = (_Float16)vb.w;
}

// ---------------------------------------------------------------------------
// Depthwise conv (k=3/5/7 across channel splits 32/16/16), same padding.
// ---------------------------------------------------------------------------
__global__ __launch_bounds__(256) void dwconv_kernel(
    const float* __restrict__ X, const float* __restrict__ w3,
    const float* __restrict__ w5, const float* __restrict__ w7,
    float* __restrict__ Y)
{
    __shared__ float tile[14 * 38];
    const int t = threadIdx.x;
    const int tx = t & 31, ty = t >> 5;
    const int x0 = blockIdx.x * 32, y0 = blockIdx.y * 8;
    const int c = blockIdx.z & 63, b = blockIdx.z >> 6;
    const float* base = X + (size_t)(b * 64 + c) * HW_FULL;

    // phase-split halo load: issue all global loads, then all LDS stores
    float hv[3];
    for (int u = 0; u < 3; ++u) {
        const int i = t + u * 256;
        float v = 0.f;
        if (i < 14 * 38) {
            int ly = i / 38, lx = i - ly * 38;
            int gy = y0 + ly - 3, gx = x0 + lx - 3;
            if (gy >= 0 && gy < 256 && gx >= 0 && gx < 256) v = base[gy * 256 + gx];
        }
        hv[u] = v;
    }
    for (int u = 0; u < 3; ++u) {
        const int i = t + u * 256;
        if (i < 14 * 38) tile[i] = hv[u];
    }
    __syncthreads();

    const float* wp; int R;
    if (c < 32)      { wp = w3 + c * 9;         R = 1; }
    else if (c < 48) { wp = w5 + (c - 32) * 25; R = 2; }
    else             { wp = w7 + (c - 48) * 49; R = 3; }
    const int K = 2 * R + 1;

    float acc = 0.f;
    for (int dy = 0; dy < K; ++dy)
        for (int dx = 0; dx < K; ++dx)
            acc += tile[(ty + 3 + dy - R) * 38 + (tx + 3 + dx - R)] * wp[dy * K + dx];

    Y[(size_t)(b * 64 + c) * HW_FULL + (y0 + ty) * 256 + (x0 + tx)] = acc;
}

// ---------------------------------------------------------------------------
// 2x2 average pool (NCHW), 256x256 -> 128x128.
// ---------------------------------------------------------------------------
__global__ __launch_bounds__(256) void avgpool_kernel(
    const float* __restrict__ X, float* __restrict__ Y)
{
    const int idx = blockIdx.x * 256 + threadIdx.x;   // over 4*64*128*128
    const int ow = idx & 127, oh = (idx >> 7) & 127, bc = idx >> 14;
    const float* p = X + (size_t)bc * HW_FULL + (oh * 2) * 256 + ow * 2;
    Y[(size_t)bc * HW_QTR + oh * 128 + ow] = 0.25f * (p[0] + p[1] + p[256] + p[257]);
}

__global__ __launch_bounds__(256) void zero_kernel(float* __restrict__ p, int n)
{
    const int i = blockIdx.x * 256 + threadIdx.x;
    if (i < n) p[i] = 0.f;
}

// ---------------------------------------------------------------------------
// C = W @ X.  W: 64x64, X: [64][HW] per batch.  WMMA f16->f32, 16x16x32.
// Double-buffered LDS pipeline; A-fragments hoisted; HW compile-time so
// row strides fold into instruction immediate offsets.
// ---------------------------------------------------------------------------
template <int HW>
__global__ __launch_bounds__(256) void gemm64_kernel(
    const float* __restrict__ X, const float* __restrict__ W,
    float* __restrict__ C)
{
    __shared__ alignas(32) _Float16 ws_[64 * 64];
    __shared__ alignas(32) _Float16 xs[2][32 * XSTR];

    const int t = threadIdx.x;
    const int b = blockIdx.y;
    X += (size_t)b * 64 * HW;
    C += (size_t)b * 64 * HW;

    for (int i = t; i < 4096; i += 256) ws_[i] = (_Float16)W[i];

    const int lane = t & 31, wv = t >> 5;
    const int lrow = lane & 15, lhi = lane >> 4;
    const int obase = (wv >> 1) * 16;
    const int pbase = (wv & 1) * 16;
    const int pix0 = blockIdx.x * 256;

    {
        float4 va, vb;
        tile_load<HW>(X, pix0, t, va, vb);
        tile_store(xs[0], t, va, vb);
    }
    __syncthreads();

    const v16h a0 = *(const v16h*)&ws_[(obase + lrow) * 64 +  0 + lhi * 16];
    const v16h a1 = *(const v16h*)&ws_[(obase + lrow) * 64 + 32 + lhi * 16];

    for (int sub = 0; sub < 8; ++sub) {
        const int cur = sub & 1;
        const int ps = pix0 + sub * 32;
        float4 va, vb;
        if (sub < 7) tile_load<HW>(X, ps + 32, t, va, vb);   // next tile in flight

        v8f acc = {};
        {
            v16h b0 = *(const v16h*)&xs[cur][(pbase + lrow) * XSTR +  0 + lhi * 16];
            acc = __builtin_amdgcn_wmma_f32_16x16x32_f16(false, a0, false, b0, (short)0, acc, false, false);
            v16h b1 = *(const v16h*)&xs[cur][(pbase + lrow) * XSTR + 32 + lhi * 16];
            acc = __builtin_amdgcn_wmma_f32_16x16x32_f16(false, a1, false, b1, (short)0, acc, false, false);
        }
        if (sub < 7) tile_store(xs[cur ^ 1], t, va, vb);
        __syncthreads();

        float* out = C + (size_t)(obase + lhi * 8) * HW + ps + pbase + lrow;
#pragma unroll
        for (int r = 0; r < 8; ++r) out[r * HW] = acc[r];   // r*HW folds to imm
    }
}

// ---------------------------------------------------------------------------
// Fused channel pooling: pooled[b,c] += sum_p (Wq@x_h)[c,p] * (Wk@y_h)[c,p].
// q/k tiles live only in accumulator registers.
// ---------------------------------------------------------------------------
__global__ __launch_bounds__(256) void qkpool_kernel(
    const float* __restrict__ Xh, const float* __restrict__ Yh,
    const float* __restrict__ Wq, const float* __restrict__ Wk,
    float* __restrict__ pooled)
{
    __shared__ alignas(32) _Float16 wqs[64 * 64];
    __shared__ alignas(32) _Float16 wks[64 * 64];
    __shared__ alignas(32) _Float16 xs[2][32 * XSTR];
    __shared__ alignas(32) _Float16 ys[2][32 * XSTR];
    __shared__ float pool_s[64];

    const int t = threadIdx.x;
    const int b = blockIdx.y;
    Xh += (size_t)b * 64 * HW_FULL;
    Yh += (size_t)b * 64 * HW_FULL;

    for (int i = t; i < 4096; i += 256) {
        wqs[i] = (_Float16)Wq[i];
        wks[i] = (_Float16)Wk[i];
    }
    if (t < 64) pool_s[t] = 0.f;

    const int lane = t & 31, wv = t >> 5;
    const int lrow = lane & 15, lhi = lane >> 4;
    const int obase = (wv >> 1) * 16;
    const int pbase = (wv & 1) * 16;
    const int pix0 = blockIdx.x * 256;

    {
        float4 va, vb, vc, vd;
        tile_load<HW_FULL>(Xh, pix0, t, va, vb);
        tile_load<HW_FULL>(Yh, pix0, t, vc, vd);
        tile_store(xs[0], t, va, vb);
        tile_store(ys[0], t, vc, vd);
    }
    __syncthreads();

    const v16h aq0 = *(const v16h*)&wqs[(obase + lrow) * 64 +  0 + lhi * 16];
    const v16h aq1 = *(const v16h*)&wqs[(obase + lrow) * 64 + 32 + lhi * 16];
    const v16h ak0 = *(const v16h*)&wks[(obase + lrow) * 64 +  0 + lhi * 16];
    const v16h ak1 = *(const v16h*)&wks[(obase + lrow) * 64 + 32 + lhi * 16];

    float prod[8];
    for (int r = 0; r < 8; ++r) prod[r] = 0.f;

    for (int sub = 0; sub < 8; ++sub) {
        const int cur = sub & 1;
        const int ps = pix0 + sub * 32;
        float4 va, vb, vc, vd;
        if (sub < 7) {
            tile_load<HW_FULL>(Xh, ps + 32, t, va, vb);
            tile_load<HW_FULL>(Yh, ps + 32, t, vc, vd);
        }

        v8f q = {}; v8f k = {};
        {
            v16h bx0 = *(const v16h*)&xs[cur][(pbase + lrow) * XSTR +  0 + lhi * 16];
            q = __builtin_amdgcn_wmma_f32_16x16x32_f16(false, aq0, false, bx0, (short)0, q, false, false);
            v16h bx1 = *(const v16h*)&xs[cur][(pbase + lrow) * XSTR + 32 + lhi * 16];
            q = __builtin_amdgcn_wmma_f32_16x16x32_f16(false, aq1, false, bx1, (short)0, q, false, false);
            v16h by0 = *(const v16h*)&ys[cur][(pbase + lrow) * XSTR +  0 + lhi * 16];
            k = __builtin_amdgcn_wmma_f32_16x16x32_f16(false, ak0, false, by0, (short)0, k, false, false);
            v16h by1 = *(const v16h*)&ys[cur][(pbase + lrow) * XSTR + 32 + lhi * 16];
            k = __builtin_amdgcn_wmma_f32_16x16x32_f16(false, ak1, false, by1, (short)0, k, false, false);
        }
        for (int r = 0; r < 8; ++r) prod[r] += q[r] * k[r];

        if (sub < 7) {
            tile_store(xs[cur ^ 1], t, va, vb);
            tile_store(ys[cur ^ 1], t, vc, vd);
        }
        __syncthreads();
    }

    // reduce over the 16 N-lanes (masks 1,2,4,8 stay within each half-wave)
    for (int m = 1; m < 16; m <<= 1)
        for (int r = 0; r < 8; ++r) prod[r] += __shfl_xor(prod[r], m);
    if (lrow == 0)
        for (int r = 0; r < 8; ++r) atomicAdd(&pool_s[obase + lhi * 8 + r], prod[r]);
    __syncthreads();
    if (t < 64) atomicAdd(&pooled[b * 64 + t], pool_s[t]);
}

// ---------------------------------------------------------------------------
// Out = P1_b @ X1 + P2_b @ X2  (per-batch 64x64 P1/P2).  Final projection.
// ---------------------------------------------------------------------------
__global__ __launch_bounds__(256) void gemm64_fin_kernel(
    const float* __restrict__ X1, const float* __restrict__ X2,
    const float* __restrict__ P1, const float* __restrict__ P2,
    float* __restrict__ Out)
{
    constexpr int HW = HW_FULL;
    __shared__ alignas(32) _Float16 w1s[64 * 64];
    __shared__ alignas(32) _Float16 w2s[64 * 64];
    __shared__ alignas(32) _Float16 x1s[2][32 * XSTR];
    __shared__ alignas(32) _Float16 x2s[2][32 * XSTR];

    const int t = threadIdx.x;
    const int b = blockIdx.y;
    X1  += (size_t)b * 64 * HW;
    X2  += (size_t)b * 64 * HW;
    Out += (size_t)b * 64 * HW;
    P1  += (size_t)b * 4096;
    P2  += (size_t)b * 4096;

    for (int i = t; i < 4096; i += 256) {
        w1s[i] = (_Float16)P1[i];
        w2s[i] = (_Float16)P2[i];
    }

    const int lane = t & 31, wv = t >> 5;
    const int lrow = lane & 15, lhi = lane >> 4;
    const int obase = (wv >> 1) * 16;
    const int pbase = (wv & 1) * 16;
    const int pix0 = blockIdx.x * 256;

    {
        float4 va, vb, vc, vd;
        tile_load<HW>(X1, pix0, t, va, vb);
        tile_load<HW>(X2, pix0, t, vc, vd);
        tile_store(x1s[0], t, va, vb);
        tile_store(x2s[0], t, vc, vd);
    }
    __syncthreads();

    const v16h a10 = *(const v16h*)&w1s[(obase + lrow) * 64 +  0 + lhi * 16];
    const v16h a11 = *(const v16h*)&w1s[(obase + lrow) * 64 + 32 + lhi * 16];
    const v16h a20 = *(const v16h*)&w2s[(obase + lrow) * 64 +  0 + lhi * 16];
    const v16h a21 = *(const v16h*)&w2s[(obase + lrow) * 64 + 32 + lhi * 16];

    for (int sub = 0; sub < 8; ++sub) {
        const int cur = sub & 1;
        const int ps = pix0 + sub * 32;
        float4 va, vb, vc, vd;
        if (sub < 7) {
            tile_load<HW>(X1, ps + 32, t, va, vb);
            tile_load<HW>(X2, ps + 32, t, vc, vd);
        }

        v8f acc = {};
        {
            v16h b10 = *(const v16h*)&x1s[cur][(pbase + lrow) * XSTR +  0 + lhi * 16];
            acc = __builtin_amdgcn_wmma_f32_16x16x32_f16(false, a10, false, b10, (short)0, acc, false, false);
            v16h b11 = *(const v16h*)&x1s[cur][(pbase + lrow) * XSTR + 32 + lhi * 16];
            acc = __builtin_amdgcn_wmma_f32_16x16x32_f16(false, a11, false, b11, (short)0, acc, false, false);
            v16h b20 = *(const v16h*)&x2s[cur][(pbase + lrow) * XSTR +  0 + lhi * 16];
            acc = __builtin_amdgcn_wmma_f32_16x16x32_f16(false, a20, false, b20, (short)0, acc, false, false);
            v16h b21 = *(const v16h*)&x2s[cur][(pbase + lrow) * XSTR + 32 + lhi * 16];
            acc = __builtin_amdgcn_wmma_f32_16x16x32_f16(false, a21, false, b21, (short)0, acc, false, false);
        }
        if (sub < 7) {
            tile_store(x1s[cur ^ 1], t, va, vb);
            tile_store(x2s[cur ^ 1], t, vc, vd);
        }
        __syncthreads();

        float* out = Out + (size_t)(obase + lhi * 8) * HW + ps + pbase + lrow;
#pragma unroll
        for (int r = 0; r < 8; ++r) out[r * HW] = acc[r];
    }
}

// ---------------------------------------------------------------------------
// high_att[b,c] = tanh(W2 @ relu(W1 @ (pooled_b / HW))).  64 thr / batch.
// ---------------------------------------------------------------------------
__global__ __launch_bounds__(64) void highatt_kernel(
    const float* __restrict__ pooled, const float* __restrict__ W1,
    const float* __restrict__ W2, float* __restrict__ hatt)
{
    __shared__ float pl[64], h1[16];
    const int b = blockIdx.x, t = threadIdx.x;
    pl[t] = pooled[b * 64 + t] * (1.f / (float)HW_FULL);
    __syncthreads();
    if (t < 16) {
        float s = 0.f;
        for (int i = 0; i < 64; ++i) s += W1[t * 64 + i] * pl[i];
        h1[t] = fmaxf(s, 0.f);
    }
    __syncthreads();
    float s = 0.f;
    for (int j = 0; j < 16; ++j) s += W2[t * 16 + j] * h1[j];
    hatt[b * 64 + t] = tanhf(s);
}

// ---------------------------------------------------------------------------
// Per (b,head): gram of l2-normalized q,k rows (8x8), softmax -> att.
// ---------------------------------------------------------------------------
__global__ __launch_bounds__(256) void lowatt_kernel(
    const float* __restrict__ QL, const float* __restrict__ KL,
    const float* __restrict__ temperature, float* __restrict__ att)
{
    __shared__ float sqk[64], sq2[8], sk2[8];
    const int bh = blockIdx.x, t = threadIdx.x;
    const int b = bh >> 3, h = bh & 7;
    const float* q = QL + (size_t)(b * 64 + h * 8) * HW_QTR;
    const float* k = KL + (size_t)(b * 64 + h * 8) * HW_QTR;

    float aqk[64], aq2[8], ak2[8];
    for (int e = 0; e < 64; ++e) aqk[e] = 0.f;
    for (int c = 0; c < 8; ++c) { aq2[c] = 0.f; ak2[c] = 0.f; }

    for (int n = t; n < HW_QTR; n += 256) {
        float qv[8], kv[8];
        for (int c = 0; c < 8; ++c) { qv[c] = q[c * HW_QTR + n]; kv[c] = k[c * HW_QTR + n]; }
        for (int c = 0; c < 8; ++c) { aq2[c] += qv[c] * qv[c]; ak2[c] += kv[c] * kv[c]; }
        for (int c = 0; c < 8; ++c)
            for (int d = 0; d < 8; ++d) aqk[c * 8 + d] += qv[c] * kv[d];
    }
    if (t < 64) sqk[t] = 0.f;
    if (t < 8) { sq2[t] = 0.f; sk2[t] = 0.f; }
    __syncthreads();
    for (int e = 0; e < 64; ++e) atomicAdd(&sqk[e], aqk[e]);
    for (int c = 0; c < 8; ++c) { atomicAdd(&sq2[c], aq2[c]); atomicAdd(&sk2[c], ak2[c]); }
    __syncthreads();

    if (t < 8) {
        const int c = t;
        const float temp = temperature[h];
        const float qn = fmaxf(sqrtf(sq2[c]), 1e-12f);
        float lg[8], mx = -3.4e38f;
        for (int d = 0; d < 8; ++d) {
            const float kn = fmaxf(sqrtf(sk2[d]), 1e-12f);
            lg[d] = sqk[c * 8 + d] / (qn * kn) * temp;
            mx = fmaxf(mx, lg[d]);
        }
        float sum = 0.f;
        for (int d = 0; d < 8; ++d) { lg[d] = __expf(lg[d] - mx); sum += lg[d]; }
        const float inv = 1.f / sum;
        for (int d = 0; d < 8; ++d)
            att[((size_t)bh * 8 + c) * 8 + d] = lg[d] * inv;
    }
}

// ---------------------------------------------------------------------------
// P1 = proj * diag(high_att),  P2 = proj @ blockdiag(att).
// ---------------------------------------------------------------------------
__global__ __launch_bounds__(256) void pmat_kernel(
    const float* __restrict__ proj, const float* __restrict__ hatt,
    const float* __restrict__ att, float* __restrict__ P1, float* __restrict__ P2)
{
    const int b = blockIdx.x, t = threadIdx.x;
    for (int e = t; e < 4096; e += 256) {
        const int o = e >> 6, i = e & 63, h = i >> 3, d = i & 7;
        P1[b * 4096 + e] = proj[e] * hatt[b * 64 + i];
        float s = 0.f;
        for (int ci = 0; ci < 8; ++ci)
            s += proj[o * 64 + h * 8 + ci] * att[((size_t)(b * 8 + h) * 8 + ci) * 8 + d];
        P2[b * 4096 + e] = s;
    }
}

// ---------------------------------------------------------------------------
extern "C" void kernel_launch(void* const* d_in, const int* in_sizes, int n_in,
                              void* d_out, int out_size, void* d_ws, size_t ws_size,
                              hipStream_t stream) {
    const float* x       = (const float*)d_in[0];
    const float* y       = (const float*)d_in[1];
    const float* hp1_w3  = (const float*)d_in[2];
    const float* hp1_w5  = (const float*)d_in[3];
    const float* hp1_w7  = (const float*)d_in[4];
    const float* hp2_w3  = (const float*)d_in[5];
    const float* hp2_w5  = (const float*)d_in[6];
    const float* hp2_w7  = (const float*)d_in[7];
    const float* qh_w    = (const float*)d_in[8];
    const float* kh_w    = (const float*)d_in[9];
    const float* vh_w    = (const float*)d_in[10];
    const float* ql_w    = (const float*)d_in[11];
    const float* kl_w    = (const float*)d_in[12];
    const float* vl_w    = (const float*)d_in[13];
    const float* proj_w  = (const float*)d_in[14];
    const float* attn_w1 = (const float*)d_in[15];
    const float* attn_w2 = (const float*)d_in[16];
    const float* temp    = (const float*)d_in[17];

    const size_t BIG = (size_t)4 * 64 * HW_FULL;   // 16.78M floats
    const size_t QTR = (size_t)4 * 64 * HW_QTR;    //  4.19M floats
    float* ws   = (float*)d_ws;
    float* bufA = ws;                 // x_h, later v_l
    float* bufB = ws + BIG;           // y_h
    float* bufC = ws + 2 * BIG;       // v_h
    float* xp   = ws + 3 * BIG;
    float* yp   = xp + QTR;
    float* qlb  = yp + QTR;
    float* klb  = qlb + QTR;
    float* sm   = klb + QTR;
    float* pooled = sm;               // 256
    float* hatt   = sm + 256;         // 256
    float* attw   = sm + 512;         // 4*8*8*8 = 2048
    float* P1     = sm + 2560;        // 4*4096
    float* P2     = sm + 2560 + 16384;

    dim3 db(256);
    // 1) depthwise convs
    dwconv_kernel<<<dim3(8, 32, 256), db, 0, stream>>>(x, hp1_w3, hp1_w5, hp1_w7, bufA);
    dwconv_kernel<<<dim3(8, 32, 256), db, 0, stream>>>(y, hp2_w3, hp2_w5, hp2_w7, bufB);
    // 2) avg pools + zero the pooled accumulator
    avgpool_kernel<<<dim3(16384), db, 0, stream>>>(x, xp);
    avgpool_kernel<<<dim3(16384), db, 0, stream>>>(y, yp);
    zero_kernel<<<dim3(1), db, 0, stream>>>(pooled, 256);
    // 3) fused q*k channel pooling (WMMA, q/k never materialized)
    qkpool_kernel<<<dim3(256, 4), db, 0, stream>>>(bufA, bufB, qh_w, kh_w, pooled);
    // 4) 1x1 conv GEMMs (WMMA)
    gemm64_kernel<HW_FULL><<<dim3(256, 4), db, 0, stream>>>(bufA, vh_w, bufC);
    gemm64_kernel<HW_FULL><<<dim3(256, 4), db, 0, stream>>>(y,    vl_w, bufA);  // x_h dead
    gemm64_kernel<HW_QTR> <<<dim3(64, 4),  db, 0, stream>>>(xp,   ql_w, qlb);
    gemm64_kernel<HW_QTR> <<<dim3(64, 4),  db, 0, stream>>>(yp,   kl_w, klb);
    // 5) channel attention MLP + low-freq attention
    highatt_kernel<<<dim3(4), dim3(64), 0, stream>>>(pooled, attn_w1, attn_w2, hatt);
    lowatt_kernel<<<dim3(32), db, 0, stream>>>(qlb, klb, temp, attw);
    // 6) fold scalars into projection, final fused WMMA GEMM into d_out
    pmat_kernel<<<dim3(4), db, 0, stream>>>(proj_w, hatt, attw, P1, P2);
    gemm64_fin_kernel<<<dim3(256, 4), db, 0, stream>>>(bufC, bufA, P1, P2, (float*)d_out);
}
